// MyModel_61933428415849
// MI455X (gfx1250) — matrix operand order repaired and checked by reading, not compiled, not run on hardware
//
#include <hip/hip_runtime.h>

// ---------------------------------------------------------------------------
// LSTM (T=512, B=64, H=800) for MI455X / gfx1250, wave32, WMMA bf16.
// Phase 0: convert x -> bf16, pack W_ih / W_hh into WMMA-fragment-ready bf16.
// Phase 1: xg = x @ W_ih^T + (b_ih + b_hh)   (big GEMM, v_wmma_f32_16x16x32_bf16)
// Phase 2: persistent-RNN kernel: W_hh slice resident in LDS, c in registers,
//          h double-buffered in global (bf16), grid barrier per timestep.
// ---------------------------------------------------------------------------

#define LSTM_T 512
#define LSTM_B 64
#define LSTM_H 800
#define LSTM_G (4 * LSTM_H)   // 3200
#define KTILES 25             // 800 / 32
#define NWG    50             // 800 / 16 hidden-unit slices

typedef __bf16 v16bf __attribute__((ext_vector_type(16)));
typedef float  v8f   __attribute__((ext_vector_type(8)));

union Frag16 {
    v16bf v;
    uint4 u[2];
};

__device__ __forceinline__ float sigmoid_f(float x) {
    return 1.0f / (1.0f + __expf(-x));
}
__device__ __forceinline__ float tanh_f(float x) {
    // tanh(x) = 1 - 2/(exp(2x)+1); exact limits at +/-inf.
    return 1.0f - 2.0f / (__expf(2.0f * x) + 1.0f);
}

#if defined(__has_builtin)
#if __has_builtin(__builtin_amdgcn_global_load_async_to_lds_b128) && \
    __has_builtin(__builtin_amdgcn_s_wait_asynccnt)
#define LSTM_ASYNC_LDS 1
#endif
#endif

#ifdef LSTM_ASYNC_LDS
// Builtin parameter types (per hipcc diagnostic): pointer-to-int4, global AS
// for the source, LDS AS for the destination.
typedef int lstm_v4i __attribute__((vector_size(4 * sizeof(int))));
typedef __attribute__((address_space(1))) lstm_v4i* lstm_gv4p;
typedef __attribute__((address_space(3))) lstm_v4i* lstm_lv4p;
#endif

// ---------------------------------------------------------------------------
// Phase 0 kernels
// ---------------------------------------------------------------------------

__global__ void lstm455_cvt_x(const float* __restrict__ in,
                              __bf16* __restrict__ out, int n) {
    int i = blockIdx.x * blockDim.x + threadIdx.x;
    if (i < n) out[i] = (__bf16)in[i];
}

// Pack W_ih (3200 x 800, row-major) into fragment-ready bf16:
// layout [ntile(200)][ktile(25)][lane(32)][elem(16)],
// value = W_ih[n][k] with n = ntile*16 + (lane&15), k = kt*32 + (lane>>4)*8 + (e&7) + (e>>3)*16.
__global__ void lstm455_pack_wih(const float* __restrict__ w,
                                 __bf16* __restrict__ wp) {
    int idx = blockIdx.x * blockDim.x + threadIdx.x;  // 200*25*512 = 2,560,000
    if (idx >= 200 * KTILES * 512) return;
    int e     = idx & 15;
    int lane  = (idx >> 4) & 31;
    int kt    = (idx >> 9) % KTILES;
    int ntile = (idx >> 9) / KTILES;
    int n = ntile * 16 + (lane & 15);
    int k = kt * 32 + (lane >> 4) * 8 + (e & 7) + (e >> 3) * 16;
    wp[idx] = (__bf16)w[(size_t)n * LSTM_H + k];
}

// Pack W_hh into per-workgroup slices:
// layout [wg(50)][gate(4)][ktile(25)][lane(32)][elem(16)].
__global__ void lstm455_pack_whh(const float* __restrict__ w,
                                 __bf16* __restrict__ wp) {
    int idx = blockIdx.x * blockDim.x + threadIdx.x;  // 50*4*25*512 = 2,560,000
    if (idx >= NWG * 4 * KTILES * 512) return;
    int e    = idx & 15;
    int lane = (idx >> 4) & 31;
    int kt   = (idx >> 9) % KTILES;
    int rest = (idx >> 9) / KTILES;
    int gate = rest & 3;
    int wg   = rest >> 2;
    int n = gate * LSTM_H + wg * 16 + (lane & 15);
    int k = kt * 32 + (lane >> 4) * 8 + (e & 7) + (e >> 3) * 16;
    wp[idx] = (__bf16)w[(size_t)n * LSTM_H + k];
}

// ---------------------------------------------------------------------------
// Phase 1: xg = x @ W_ih^T + bias. grid = (512, 50), block = 128 (4 waves);
// each wave computes a 16x64 C strip. K loop fully unrolled: accumulators stay
// pinned in registers (no loop-PHI copies), WMMAs chain without hazard NOPs.
// ---------------------------------------------------------------------------

__global__ void lstm455_xg_gemm(const __bf16* __restrict__ xb,     // [32768][800]
                                const __bf16* __restrict__ wpack,  // fragment-ready
                                const float* __restrict__ bih,
                                const float* __restrict__ bhh,
                                float* __restrict__ xg) {          // [32768][3200]
    const int mblk = blockIdx.x;
    const int nblk = blockIdx.y;
    const int wave = threadIdx.x >> 5;
    const int lane = threadIdx.x & 31;
    const int lrow = lane & 15;
    const int lhalf = lane >> 4;
    const int row0 = mblk * 64 + wave * 16;

    v8f acc[4] = {};

    // A fragment (16-bit 16x32): lane -> row (row0+lrow); two b128 loads at
    // K offsets lhalf*8 and lhalf*8+16 within each K-tile of 32.
    const __bf16* arow = xb + (size_t)(row0 + lrow) * LSTM_H + lhalf * 8;
    const __bf16* bbase =
        wpack + ((((size_t)nblk * 4) * KTILES) * 32 + lane) * 16;

    #pragma unroll
    for (int kt = 0; kt < KTILES; ++kt) {
        Frag16 a;
        a.u[0] = *(const uint4*)(arow + kt * 32);
        a.u[1] = *(const uint4*)(arow + kt * 32 + 16);
        #pragma unroll
        for (int nt = 0; nt < 4; ++nt) {
            const __bf16* bp = bbase + ((size_t)nt * KTILES + kt) * 512;
            Frag16 b;
            b.u[0] = *(const uint4*)(bp);
            b.u[1] = *(const uint4*)(bp + 8);
            acc[nt] = __builtin_amdgcn_wmma_f32_16x16x32_bf16(
                false, a.v, false, b.v, (short)0, acc[nt], false, false);
        }
    }

    // C layout: lane&15 -> N, lane>>4 selects M rows {0..7 | 8..15}, vgpr e -> M.
    #pragma unroll
    for (int nt = 0; nt < 4; ++nt) {
        const int col = nblk * 64 + nt * 16 + lrow;
        const float bias = bih[col] + bhh[col];
        #pragma unroll
        for (int e = 0; e < 8; ++e) {
            const int r = row0 + lhalf * 8 + e;
            xg[(size_t)r * LSTM_G + col] = acc[nt][e] + bias;
        }
    }
}

// ---------------------------------------------------------------------------
// Phase 2: persistent recurrence. grid = 50 WGs x 128 threads, 102400 B LDS.
// ---------------------------------------------------------------------------

__global__ void lstm455_recur(const float* __restrict__ xg,        // [T][64][3200]
                              const __bf16* __restrict__ whh_pack, // [50][4][25][32][16]
                              __bf16* __restrict__ hbuf,           // [2][64][800]
                              float* __restrict__ out,             // [T][64][800]
                              unsigned* __restrict__ bar) {
    extern __shared__ __align__(16) unsigned char smem[];  // 4*25*32*32B = 102400

    const int wg   = blockIdx.x;         // 0..49
    const int wave = threadIdx.x >> 5;   // 0..3 -> rows wave*16..wave*16+15
    const int lane = threadIdx.x & 31;
    const int lrow = lane & 15;
    const int lhalf = lane >> 4;
    const int unit = wg * 16 + lrow;     // hidden unit this lane owns (per C col)

    // Load this WG's W_hh slice (102400 B) into LDS once; resident all 512
    // steps. Prefer the gfx1250 async-to-LDS path (bypasses VGPRs, ASYNCcnt).
    {
        const __bf16* src = whh_pack + (size_t)wg * 4 * KTILES * 512;
#ifdef LSTM_ASYNC_LDS
        char* srcnc = (char*)src;  // drop const, then addrspace-cast
        __attribute__((address_space(1))) char* gsrc =
            (__attribute__((address_space(1))) char*)srcnc;
        __attribute__((address_space(3))) char* ldst =
            (__attribute__((address_space(3))) char*)smem;
        for (int i = threadIdx.x * 16; i < 102400; i += 128 * 16) {
            __builtin_amdgcn_global_load_async_to_lds_b128(
                (lstm_gv4p)(gsrc + i), (lstm_lv4p)(ldst + i), 0, 0);
        }
        __builtin_amdgcn_s_wait_asynccnt(0);
#else
        const uint4* s4 = (const uint4*)src;
        uint4* dst = (uint4*)smem;
        for (int i = threadIdx.x; i < 102400 / 16; i += 128) dst[i] = s4[i];
#endif
    }
    __syncthreads();

    float cst[8];
    #pragma unroll
    for (int e = 0; e < 8; ++e) cst[e] = 0.0f;

    for (int t = 0; t < LSTM_T; ++t) {
        const __bf16* hread = hbuf + (size_t)(t & 1) * LSTM_B * LSTM_H;
        __bf16* hwrite      = hbuf + (size_t)((t + 1) & 1) * LSTM_B * LSTM_H;

        // Preload this step's xg gate pre-activations so the L2 round-trip
        // overlaps the WMMA stream (xg is read-only; no barrier dependency).
        float xv[4][8];
        #pragma unroll
        for (int e = 0; e < 8; ++e) {
            const int r = wave * 16 + lhalf * 8 + e;
            const float* xr = xg + ((size_t)t * LSTM_B + r) * LSTM_G;
            #pragma unroll
            for (int g = 0; g < 4; ++g) xv[g][e] = xr[g * LSTM_H + unit];
        }

        v8f acc[4] = {};
        const __bf16* arow =
            hread + (size_t)(wave * 16 + lrow) * LSTM_H + lhalf * 8;

        #pragma unroll
        for (int kt = 0; kt < KTILES; ++kt) {
            Frag16 a;
            a.u[0] = *(const uint4*)(arow + kt * 32);
            a.u[1] = *(const uint4*)(arow + kt * 32 + 16);
            #pragma unroll
            for (int g = 0; g < 4; ++g) {
                const uint4* bp =
                    (const uint4*)(smem + (((g * KTILES + kt) * 32 + lane) << 5));
                Frag16 b;
                b.u[0] = bp[0];
                b.u[1] = bp[1];
                acc[g] = __builtin_amdgcn_wmma_f32_16x16x32_bf16(
                    false, a.v, false, b.v, (short)0, acc[g], false, false);
            }
        }

        // Elementwise LSTM cell update; each lane owns 8 (row, unit) pairs.
        #pragma unroll
        for (int e = 0; e < 8; ++e) {
            const int r = wave * 16 + lhalf * 8 + e;
            float iv = sigmoid_f(acc[0][e] + xv[0][e]);
            float fv = sigmoid_f(acc[1][e] + xv[1][e]);
            float gv = tanh_f   (acc[2][e] + xv[2][e]);
            float ov = sigmoid_f(acc[3][e] + xv[3][e]);
            float cv = fv * cst[e] + iv * gv;
            cst[e] = cv;
            const float hv = ov * tanh_f(cv);
            out[((size_t)t * LSTM_B + r) * LSTM_H + unit] = hv;
            hwrite[(size_t)r * LSTM_H + unit] = (__bf16)hv;
        }

        // Grid-wide barrier (monotonic counter, device-scope fences so h
        // written on other WGPs is visible past the WGP$).
        __syncthreads();
        if (threadIdx.x == 0) {
            __builtin_amdgcn_fence(__ATOMIC_RELEASE, "agent");
            atomicAdd(bar, 1u);
            const unsigned target = (unsigned)(t + 1) * (unsigned)NWG;
            while (__hip_atomic_load(bar, __ATOMIC_RELAXED,
                                     __HIP_MEMORY_SCOPE_AGENT) < target) {
                __builtin_amdgcn_s_sleep(2);
            }
        }
        __syncthreads();
        __builtin_amdgcn_fence(__ATOMIC_ACQUIRE, "agent");
    }
}

// ---------------------------------------------------------------------------
// Workspace layout (bytes):
//   [0,256)              barrier counter (zeroed each call)
//   [256, +204800)       h double buffer, bf16 (zeroed each call)
//   [205312, +5120000)   W_ih packed bf16
//   [+5120000)           W_hh packed bf16
//   then x bf16 (52 MB), then xg fp32 (419 MB)
// ---------------------------------------------------------------------------

extern "C" void kernel_launch(void* const* d_in, const int* in_sizes, int n_in,
                              void* d_out, int out_size, void* d_ws, size_t ws_size,
                              hipStream_t stream) {
    const float* x    = (const float*)d_in[0];
    const float* W_ih = (const float*)d_in[1];
    const float* W_hh = (const float*)d_in[2];
    const float* b_ih = (const float*)d_in[3];
    const float* b_hh = (const float*)d_in[4];
    float* out = (float*)d_out;

    char* ws = (char*)d_ws;
    const size_t OFF_BAR  = 0;
    const size_t OFF_H    = 256;                                    // 204800 B
    const size_t OFF_WIH  = 205312;                                 // 5120000 B
    const size_t OFF_WHH  = OFF_WIH + 5120000;                      // 5120000 B
    const size_t OFF_XB   = OFF_WHH + 5120000;                      // 52428800 B
    const size_t OFF_XG   = OFF_XB + (size_t)LSTM_T * LSTM_B * LSTM_H * 2;

    unsigned* bar     = (unsigned*)(ws + OFF_BAR);
    __bf16*   hbuf    = (__bf16*)(ws + OFF_H);
    __bf16*   wih_pk  = (__bf16*)(ws + OFF_WIH);
    __bf16*   whh_pk  = (__bf16*)(ws + OFF_WHH);
    __bf16*   x_bf    = (__bf16*)(ws + OFF_XB);
    float*    xg      = (float*)(ws + OFF_XG);

    (void)hipMemsetAsync(bar, 0, 256, stream);
    (void)hipMemsetAsync(hbuf, 0, 2 * LSTM_B * LSTM_H * 2, stream);

    // Phase 0
    {
        const int n = LSTM_T * LSTM_B * LSTM_H;  // 26,214,400
        lstm455_cvt_x<<<(n + 255) / 256, 256, 0, stream>>>(x, x_bf, n);
        lstm455_pack_wih<<<(200 * KTILES * 512) / 256, 256, 0, stream>>>(W_ih, wih_pk);
        lstm455_pack_whh<<<(NWG * 4 * KTILES * 512) / 256, 256, 0, stream>>>(W_hh, whh_pk);
    }

    // Phase 1: input-projection GEMM
    lstm455_xg_gemm<<<dim3(512, 50), 128, 0, stream>>>(x_bf, wih_pk, b_ih, b_hh, xg);

    // Phase 2: persistent recurrence (50 WGs, W_hh slice resident in LDS)
    lstm455_recur<<<NWG, 128, 102400, stream>>>(xg, whh_pk, hbuf, out, bar);
}
